// GNNBlock_72353019069071
// MI455X (gfx1250) — matrix operand order repaired and checked by reading, not compiled, not run on hardware
//
#include <hip/hip_runtime.h>
#include <hip/hip_bf16.h>
#include <stdint.h>

#define HIDDEN 128
#define LN_EPS 1e-5f
#define TPAD   136   // padded LDS row stride (bf16 elems): 272B -> conflict-free frag loads

typedef __bf16 bf16_t;
typedef __attribute__((ext_vector_type(16))) __bf16 v16bf;
typedef __attribute__((ext_vector_type(8)))  __bf16 v8bf;
typedef __attribute__((ext_vector_type(8)))  float  v8f;

__device__ __forceinline__ unsigned short f2bfu(float f) {
    union { float f; unsigned u; } c; c.f = f;
    unsigned r = c.u + 0x7FFFu + ((c.u >> 16) & 1u);   // round-to-nearest-even
    return (unsigned short)(r >> 16);
}
__device__ __forceinline__ bf16_t us2bf(unsigned short u) {
    union { unsigned short u; bf16_t b; } c; c.u = u; return c.b;
}

// ---------------------------------------------------------------------------
// Kernel 1: agg = x (copy), and convert W1/W2 to bf16 (L2-resident weights)
// ---------------------------------------------------------------------------
__global__ void prep_kernel(const float4* __restrict__ x4,
                            const float*  __restrict__ W1,
                            const float*  __restrict__ W2,
                            float4* __restrict__ agg4,
                            bf16_t* __restrict__ w1bf,
                            bf16_t* __restrict__ w2bf,
                            long long n4) {
    long long i      = (long long)blockIdx.x * blockDim.x + threadIdx.x;
    long long stride = (long long)gridDim.x * blockDim.x;
    for (long long e = i; e < n4; e += stride)
        agg4[e] = x4[e];
    for (long long e = i; e < (long long)HIDDEN * HIDDEN; e += stride) {
        w1bf[e] = us2bf(f2bfu(W1[e]));
        w2bf[e] = us2bf(f2bfu(W2[e]));
    }
}

// ---------------------------------------------------------------------------
// Kernel 2: agg[dst] += x[src] over edges. One 32-lane wave per edge,
// float4 gather per lane, 4x hardware f32 atomic adds (no return -> STOREcnt).
// ---------------------------------------------------------------------------
__global__ void scatter_kernel(const float* __restrict__ x,
                               const int*   __restrict__ ei,
                               float* __restrict__ agg,
                               int nEdges) {
    int gid  = blockIdx.x * blockDim.x + threadIdx.x;
    int e    = gid >> 5;
    if (e >= nEdges) return;
    int lane = gid & 31;
    int src  = ei[e];
    int dst  = ei[nEdges + e];
    const float4 v = *(const float4*)(x + (size_t)src * HIDDEN + lane * 4);
    float* p = agg + (size_t)dst * HIDDEN + lane * 4;
    unsafeAtomicAdd(p + 0, v.x);
    unsafeAtomicAdd(p + 1, v.y);
    unsafeAtomicAdd(p + 2, v.z);
    unsafeAtomicAdd(p + 3, v.w);
}

// ---------------------------------------------------------------------------
// Kernel 3: fused  relu(agg@W1^T+b1)@W2^T+b2 -> LayerNorm -> ReLU -> +x
// 8 waves/block, 16 node-rows per wave, bf16 WMMA 16x16x32, fp32 accum.
// ---------------------------------------------------------------------------
__global__ __launch_bounds__(256) void mlp_kernel(
    const float*  __restrict__ x,
    const float*  __restrict__ agg,
    const bf16_t* __restrict__ w1bf,
    const bf16_t* __restrict__ w2bf,
    const float*  __restrict__ b1,
    const float*  __restrict__ b2,
    const float*  __restrict__ gamma,
    const float*  __restrict__ beta,
    float* __restrict__ out,
    int nNodes) {

    __shared__ bf16_t tile[8][16 * TPAD];   // per-wave 16x128 bf16 tile (padded)

    const int lane    = threadIdx.x & 31;
    const int wave    = threadIdx.x >> 5;
    const int rowbase = blockIdx.x * 128 + wave * 16;
    if (rowbase >= nNodes) return;          // wave-uniform exit, EXEC stays full

    bf16_t* my = &tile[wave][0];
    const int r    = lane & 15;             // 0..15 within half
    const int half = lane >> 4;             // K-half selector per ISA A/B layout

    // ---- stage contiguous 16x128 fp32 agg tile -> bf16 LDS (coalesced) ----
    #pragma unroll
    for (int it = 0; it < 16; ++it) {
        int grow = rowbase + it;
        if (grow >= nNodes) grow = nNodes - 1;      // clamp: keep loads valid
        const float4 v = *(const float4*)(agg + (size_t)grow * HIDDEN + lane * 4);
        ushort4 p;
        p.x = f2bfu(v.x); p.y = f2bfu(v.y); p.z = f2bfu(v.z); p.w = f2bfu(v.w);
        *(ushort4*)(my + it * TPAD + lane * 4) = p;
    }

    // ---- A fragments for GEMM1: 16x32 bf16, ISA lane layout ----
    v16bf a1[4];
    #pragma unroll
    for (int kb = 0; kb < 4; ++kb) {
        const bf16_t* base = my + r * TPAD + kb * 32 + half * 8;
        v8bf lo = *(const v8bf*)(base);          // K = kb*32 + half*8 + [0..7]
        v8bf hi = *(const v8bf*)(base + 16);     // K = kb*32 + 16 + half*8 + [0..7]
        a1[kb] = __builtin_shufflevector(lo, hi, 0,1,2,3,4,5,6,7,8,9,10,11,12,13,14,15);
    }

    // ---- GEMM1 (+b1, ReLU) -> same LDS tile as bf16 h1 ----
    #pragma unroll
    for (int nt = 0; nt < 8; ++nt) {
        v8f acc = {0.f,0.f,0.f,0.f,0.f,0.f,0.f,0.f};
        #pragma unroll
        for (int kb = 0; kb < 4; ++kb) {
            const bf16_t* wb = w1bf + (size_t)(nt * 16 + r) * HIDDEN + kb * 32 + half * 8;
            v8bf lo = *(const v8bf*)(wb);        // B col = nt*16+r, K = kb*32+half*8+[0..7]
            v8bf hi = *(const v8bf*)(wb + 16);   //                 K = kb*32+16+half*8+[0..7]
            v16bf bfrag = __builtin_shufflevector(lo, hi, 0,1,2,3,4,5,6,7,8,9,10,11,12,13,14,15);
            acc = __builtin_amdgcn_wmma_f32_16x16x32_bf16(
                      false, a1[kb], false, bfrag, (short)0, acc, false, false);
        }
        const float bias = b1[nt * 16 + r];
        #pragma unroll
        for (int g = 0; g < 8; ++g) {
            float h = fmaxf(acc[g] + bias, 0.f); // C/D layout: M=g+half*8, N=r
            my[(g + half * 8) * TPAD + nt * 16 + r] = us2bf(f2bfu(h));
        }
    }

    // ---- A fragments for GEMM2 from h1 in LDS ----
    v16bf a2[4];
    #pragma unroll
    for (int kb = 0; kb < 4; ++kb) {
        const bf16_t* base = my + r * TPAD + kb * 32 + half * 8;
        v8bf lo = *(const v8bf*)(base);
        v8bf hi = *(const v8bf*)(base + 16);
        a2[kb] = __builtin_shufflevector(lo, hi, 0,1,2,3,4,5,6,7,8,9,10,11,12,13,14,15);
    }

    // ---- GEMM2, keep all 8 N-tiles of accumulators live for LayerNorm ----
    v8f acc2[8];
    #pragma unroll
    for (int nt = 0; nt < 8; ++nt) {
        v8f acc = {0.f,0.f,0.f,0.f,0.f,0.f,0.f,0.f};
        #pragma unroll
        for (int kb = 0; kb < 4; ++kb) {
            const bf16_t* wb = w2bf + (size_t)(nt * 16 + r) * HIDDEN + kb * 32 + half * 8;
            v8bf lo = *(const v8bf*)(wb);
            v8bf hi = *(const v8bf*)(wb + 16);
            v16bf bfrag = __builtin_shufflevector(lo, hi, 0,1,2,3,4,5,6,7,8,9,10,11,12,13,14,15);
            acc = __builtin_amdgcn_wmma_f32_16x16x32_bf16(
                      false, a2[kb], false, bfrag, (short)0, acc, false, false);
        }
        const float bias = b2[nt * 16 + r];
        #pragma unroll
        for (int g = 0; g < 8; ++g) acc[g] += bias;
        acc2[nt] = acc;
    }

    // ---- LayerNorm stats fully in registers: reduce over 8 tiles, then
    //      shfl_xor across the 16 lanes of each half (masks stay in-half) ----
    float s[8], q[8];
    #pragma unroll
    for (int g = 0; g < 8; ++g) { s[g] = 0.f; q[g] = 0.f; }
    #pragma unroll
    for (int nt = 0; nt < 8; ++nt) {
        #pragma unroll
        for (int g = 0; g < 8; ++g) {
            float h = acc2[nt][g];
            s[g] += h;
            q[g] += h * h;
        }
    }
    #pragma unroll
    for (int g = 0; g < 8; ++g) {
        #pragma unroll
        for (int m = 1; m < 16; m <<= 1) {
            s[g] += __shfl_xor(s[g], m, 32);
            q[g] += __shfl_xor(q[g], m, 32);
        }
    }
    float mu[8], rstd[8];
    #pragma unroll
    for (int g = 0; g < 8; ++g) {
        mu[g] = s[g] * (1.0f / HIDDEN);
        float var = q[g] * (1.0f / HIDDEN) - mu[g] * mu[g];
        rstd[g] = rsqrtf(var + LN_EPS);
    }

    // ---- epilogue: gamma/beta, ReLU, residual, store ----
    #pragma unroll
    for (int nt = 0; nt < 8; ++nt) {
        const int col = nt * 16 + r;
        const float gm = gamma[col];
        const float bt = beta[col];
        #pragma unroll
        for (int g = 0; g < 8; ++g) {
            const int row = rowbase + g + half * 8;
            if (row < nNodes) {
                float v = (acc2[nt][g] - mu[g]) * rstd[g] * gm + bt;
                v = fmaxf(v, 0.f);
                out[(size_t)row * HIDDEN + col] = x[(size_t)row * HIDDEN + col] + v;
            }
        }
    }
}

// ---------------------------------------------------------------------------
extern "C" void kernel_launch(void* const* d_in, const int* in_sizes, int n_in,
                              void* d_out, int out_size, void* d_ws, size_t ws_size,
                              hipStream_t stream) {
    const float* x     = (const float*)d_in[0];
    const int*   ei    = (const int*)d_in[1];
    const float* W1    = (const float*)d_in[2];
    const float* b1    = (const float*)d_in[3];
    const float* W2    = (const float*)d_in[4];
    const float* b2    = (const float*)d_in[5];
    const float* gamma = (const float*)d_in[6];
    const float* beta  = (const float*)d_in[7];
    float* out = (float*)d_out;

    const int nNodes = in_sizes[0] / HIDDEN;
    const int nEdges = in_sizes[1] / 2;

    // workspace layout: [agg fp32 | W1 bf16 | W2 bf16]
    float* agg = (float*)d_ws;
    size_t aggBytes = (size_t)nNodes * HIDDEN * sizeof(float);
    bf16_t* w1bf = (bf16_t*)((char*)d_ws + aggBytes);
    bf16_t* w2bf = w1bf + (size_t)HIDDEN * HIDDEN;

    long long n4 = (long long)nNodes * HIDDEN / 4;
    int blocks1 = (int)((n4 + 255) / 256);
    prep_kernel<<<blocks1, 256, 0, stream>>>(
        (const float4*)x, W1, W2, (float4*)agg, w1bf, w2bf, n4);

    long long tot = (long long)nEdges * 32;
    int blocks2 = (int)((tot + 255) / 256);
    scatter_kernel<<<blocks2, 256, 0, stream>>>(x, ei, agg, nEdges);

    int blocks3 = (nNodes + 127) / 128;
    mlp_kernel<<<blocks3, 256, 0, stream>>>(
        x, agg, w1bf, w2bf, b1, b2, gamma, beta, out, nNodes);
}
